// GCNGraphEncoder_18940805775859
// MI455X (gfx1250) — matrix operand order repaired and checked by reading, not compiled, not run on hardware
//
#include <hip/hip_runtime.h>

typedef __attribute__((ext_vector_type(16))) __bf16 v16bf;
typedef __attribute__((ext_vector_type(8)))  __bf16 v8bf;
typedef __attribute__((ext_vector_type(8)))  float  v8f;

#define NN   50000    // nodes
#define NE   800000   // edges
#define NG   256      // graphs
#define DF   128      // in features
#define DH   128      // hidden
#define DHID 256
#define DOUT 128

// ---------------- elementwise / utility kernels ----------------

__global__ void k_fill(float* __restrict__ p, float v, int n) {
  int i = blockIdx.x * blockDim.x + threadIdx.x;
  if (i < n) p[i] = v;
}

// Pack fp32 weight [K][Ncols] into bf16 WMMA B-fragment order:
// packed[((kt*(Ncols/16)+ct)*32 + lane)*16 + i] = W[(kt*32 + (lane>>4)*16 + i)*Ncols + ct*16 + (lane&15)]
__global__ void k_pack_w_bf16(const float* __restrict__ W, __bf16* __restrict__ P,
                              int K, int Ncols) {
  int idx = blockIdx.x * blockDim.x + threadIdx.x;
  int total = K * Ncols;
  if (idx >= total) return;
  int i    = idx & 15;
  int lane = (idx >> 4) & 31;
  int rest = idx >> 9;
  int nct  = Ncols >> 4;
  int ct   = rest % nct;
  int kt   = rest / nct;
  int k = kt * 32 + (lane >> 4) * 16 + i;
  int n = ct * 16 + (lane & 15);
  P[idx] = (__bf16)W[(size_t)k * Ncols + n];
}

__global__ void k_deg_edges(const int* __restrict__ dst, float* __restrict__ deg, int e) {
  int i = blockIdx.x * blockDim.x + threadIdx.x;
  if (i < e) atomicAdd(&deg[dst[i]], 1.0f);
}

__global__ void k_rsqrt_inplace(float* __restrict__ p, int n) {
  int i = blockIdx.x * blockDim.x + threadIdx.x;
  if (i < n) p[i] = rsqrtf(fmaxf(p[i], 1.0f));
}

// acc[i][f] = h0[i][f] * dinv[i]^2   (GCN self-loop contribution)
__global__ void k_gcn_self(const float* __restrict__ h0, const float* __restrict__ dinv,
                           float* __restrict__ acc) {
  int idx = blockIdx.x * blockDim.x + threadIdx.x;
  if (idx >= NN * DH) return;
  int node = idx >> 7;            // DH == 128
  float di = dinv[node];
  acc[idx] = h0[idx] * di * di;
}

// per-edge normalized scatter: one wave-slice of 32 lanes per edge, float4 per lane
__global__ void k_gcn_scatter(const int* __restrict__ src, const int* __restrict__ dst,
                              const float* __restrict__ dinv, const float* __restrict__ h0,
                              float* __restrict__ acc) {
  long long t = (long long)blockIdx.x * blockDim.x + threadIdx.x;
  int e = (int)(t >> 5);
  int part = (int)(t & 31);
  if (e >= NE) return;
  int s = src[e], d = dst[e];
  float w = dinv[s] * dinv[d];
  const float4 v = ((const float4*)(h0 + (size_t)s * DH))[part];
  float* ad = acc + (size_t)d * DH + part * 4;
  atomicAdd(ad + 0, v.x * w);
  atomicAdd(ad + 1, v.y * w);
  atomicAdd(ad + 2, v.z * w);
  atomicAdd(ad + 3, v.w * w);
}

__global__ void k_gin_scatter(const int* __restrict__ src, const int* __restrict__ dst,
                              const float* __restrict__ h, float* __restrict__ acc) {
  long long t = (long long)blockIdx.x * blockDim.x + threadIdx.x;
  int e = (int)(t >> 5);
  int part = (int)(t & 31);
  if (e >= NE) return;
  int s = src[e], d = dst[e];
  const float4 v = ((const float4*)(h + (size_t)s * DH))[part];
  float* ad = acc + (size_t)d * DH + part * 4;
  atomicAdd(ad + 0, v.x);
  atomicAdd(ad + 1, v.y);
  atomicAdd(ad + 2, v.z);
  atomicAdd(ad + 3, v.w);
}

__global__ void k_bias_relu(const float* __restrict__ in, const float* __restrict__ bias,
                            float* __restrict__ out, int n, int hdim) {
  int i = blockIdx.x * blockDim.x + threadIdx.x;
  if (i < n) out[i] = fmaxf(in[i] + bias[i & (hdim - 1)], 0.0f);
}

__global__ void k_copy4(const float4* __restrict__ in, float4* __restrict__ out, int n4) {
  int i = blockIdx.x * blockDim.x + threadIdx.x;
  if (i < n4) out[i] = in[i];
}

__global__ void k_pool_scatter(const float* __restrict__ h, const int* __restrict__ batch,
                               float* __restrict__ s, float* __restrict__ cnt) {
  long long t = (long long)blockIdx.x * blockDim.x + threadIdx.x;
  int node = (int)(t >> 5);
  int part = (int)(t & 31);
  if (node >= NN) return;
  int g = batch[node];
  const float4 v = ((const float4*)(h + (size_t)node * DH))[part];
  float* sd = s + (size_t)g * DH + part * 4;
  atomicAdd(sd + 0, v.x);
  atomicAdd(sd + 1, v.y);
  atomicAdd(sd + 2, v.z);
  atomicAdd(sd + 3, v.w);
  if (part == 0) atomicAdd(&cnt[g], 1.0f);
}

__global__ void k_pool_mean(const float* __restrict__ s, const float* __restrict__ cnt,
                            float* __restrict__ m) {
  int idx = blockIdx.x * blockDim.x + threadIdx.x;
  if (idx >= NG * DH) return;
  int g = idx >> 7;
  m[idx] = s[idx] / fmaxf(cnt[g], 1.0f);
}

// ---------------- WMMA GEMM: out[M][Ncols] = op(A[M][K] @ Wpacked + bias) ----------------
// 256 threads = 8 waves; block tile = 16 rows x 128 cols; wave w owns a 16x16 D tile.
// The fp32 A tile (16 x K) is staged ONCE into LDS as bf16 by all 8 waves cooperatively,
// then each wave reads its A fragments with 16-byte ds loads. B fragments come from the
// pre-swizzled bf16 weight buffer as one contiguous 32-byte run per lane, with
// global_prefetch of the next K-tile overlapping the WMMA.
// Requires: M % 16 == 0, K % 32 == 0 (K <= 256), Ncols % 128 == 0 via grid.y.
__global__ void k_gemm_bf16(const float* __restrict__ A, const __bf16* __restrict__ Wp,
                            const float* __restrict__ bias, float* __restrict__ out,
                            int M, int K, int Ncols, int relu) {
  __shared__ __bf16 sA[16 * 256];   // 8 KB max (K <= 256)

  const int tid  = threadIdx.x;
  const int lane = tid & 31;
  const int wave = tid >> 5;
  const int rowBase = blockIdx.x * 16;
  const int colTile = blockIdx.y * 8 + wave;      // global 16-col tile index
  const int colBase = colTile * 16;
  const int nct = Ncols >> 4;

  // ---- cooperative stage: 16 x K fp32 -> bf16 into LDS (row-major, K contiguous) ----
  const int total = 16 * K;
  for (int idx = tid; idx < total; idx += 256) {
    int r = idx / K;
    int k = idx - r * K;
    sA[idx] = (__bf16)A[(size_t)(rowBase + r) * K + k];
  }
  __syncthreads();

  const int arow = (lane & 15) * K;
  const int half = lane >> 4;

  v8f c = {};
  for (int k0 = 0; k0 < K; k0 += 32) {
    // A fragment from LDS: two contiguous 16B runs per lane
    const int kA = k0 + half * 16;
    v8bf a0 = *(const v8bf*)(sA + arow + kA);
    v8bf a1 = *(const v8bf*)(sA + arow + kA + 8);
    v16bf a = __builtin_shufflevector(a0, a1, 0, 1, 2, 3, 4, 5, 6, 7,
                                              8, 9, 10, 11, 12, 13, 14, 15);
    // B fragment: one contiguous 32B run per lane from the pre-swizzled weights
    const int kt = k0 >> 5;
    const __bf16* bp = Wp + ((size_t)(kt * nct + colTile) * 32 + lane) * 16;
    v16bf b = *(const v16bf*)bp;
    if (k0 + 32 < K)  // prefetch next K-tile of B into cache (global_prefetch_b8)
      __builtin_prefetch((const void*)(bp + (size_t)nct * 512), 0, 0);

    c = __builtin_amdgcn_wmma_f32_16x16x32_bf16(false, a, false, b, (short)0, c,
                                                false, false);
  }

  const int ocol = colBase + (lane & 15);
  const float bv = bias ? bias[ocol] : 0.0f;
#pragma unroll
  for (int r = 0; r < 8; ++r) {
    int orow = rowBase + r + 8 * half;
    float v = c[r] + bv;
    if (relu) v = fmaxf(v, 0.0f);
    out[(size_t)orow * Ncols + ocol] = v;
  }
}

// ---------------- host-side launch ----------------

extern "C" void kernel_launch(void* const* d_in, const int* in_sizes, int n_in,
                              void* d_out, int out_size, void* d_ws, size_t ws_size,
                              hipStream_t stream) {
  const float* x    = (const float*)d_in[0];
  const int*   ei   = (const int*)d_in[1];      // [2][E]
  const int*   batch= (const int*)d_in[2];
  const float* W0   = (const float*)d_in[3];
  const float* b0   = (const float*)d_in[4];
  const float* Wg1  = (const float*)d_in[5];
  const float* bg1  = (const float*)d_in[6];
  const float* Wg2  = (const float*)d_in[7];
  const float* bg2  = (const float*)d_in[8];
  const float* Wh1  = (const float*)d_in[9];
  const float* bh1  = (const float*)d_in[10];
  const float* Wh2  = (const float*)d_in[11];
  const float* bh2  = (const float*)d_in[12];
  const int* esrc = ei;
  const int* edst = ei + NE;

  char* wp = (char*)d_ws;
  auto alloc = [&](size_t bytes) -> void* {
    void* r = (void*)wp;
    wp += (bytes + 255) & ~(size_t)255;
    return r;
  };
  float*  bufA  = (float*)alloc((size_t)NN * DH * 4);
  float*  bufB  = (float*)alloc((size_t)NN * DH * 4);
  float*  dinv  = (float*)alloc((size_t)NN * 4);
  float*  spool = (float*)alloc((size_t)NG * DH * 4);
  float*  cnt   = (float*)alloc((size_t)NG * 4);
  float*  mbuf  = (float*)alloc((size_t)NG * DH * 4);
  float*  gbuf  = (float*)alloc((size_t)NG * DHID * 4);
  __bf16* W0b   = (__bf16*)alloc((size_t)DF * DH * 2);
  __bf16* Wg1b  = (__bf16*)alloc((size_t)DH * DH * 2);
  __bf16* Wg2b  = (__bf16*)alloc((size_t)DH * DH * 2);
  __bf16* Wh1b  = (__bf16*)alloc((size_t)DH * DHID * 2);
  __bf16* Wh2b  = (__bf16*)alloc((size_t)DHID * DOUT * 2);

  const int TB = 256;
  auto blks = [](long long n, int tb) { return (unsigned)((n + tb - 1) / tb); };

  // 1) pack weights to bf16 fragment layout
  k_pack_w_bf16<<<blks(DF * DH, TB), TB, 0, stream>>>(W0,  W0b,  DF,  DH);
  k_pack_w_bf16<<<blks(DH * DH, TB), TB, 0, stream>>>(Wg1, Wg1b, DH,  DH);
  k_pack_w_bf16<<<blks(DH * DH, TB), TB, 0, stream>>>(Wg2, Wg2b, DH,  DH);
  k_pack_w_bf16<<<blks(DH * DHID, TB), TB, 0, stream>>>(Wh1, Wh1b, DH,  DHID);
  k_pack_w_bf16<<<blks((long long)DHID * DOUT, TB), TB, 0, stream>>>(Wh2, Wh2b, DHID, DOUT);

  // 2) degrees (self-loop => init 1.0), then dinv = rsqrt(deg)
  k_fill<<<blks(NN, TB), TB, 0, stream>>>(dinv, 1.0f, NN);
  k_deg_edges<<<blks(NE, TB), TB, 0, stream>>>(edst, dinv, NE);
  k_rsqrt_inplace<<<blks(NN, TB), TB, 0, stream>>>(dinv, NN);

  // 3) h0 = x @ W0  (bias deferred until after aggregation)
  k_gemm_bf16<<<dim3(NN / 16, DH / 128), TB, 0, stream>>>(x, W0b, nullptr, bufA,
                                                          NN, DF, DH, 0);

  // 4) GCN aggregation: bufB = h0*dinv^2 + sum_e norm*h0[src], then relu(+b0) -> bufA
  k_gcn_self<<<blks((long long)NN * DH, TB), TB, 0, stream>>>(bufA, dinv, bufB);
  k_gcn_scatter<<<blks((long long)NE * 32, TB), TB, 0, stream>>>(esrc, edst, dinv, bufA, bufB);
  k_bias_relu<<<blks((long long)NN * DH, TB), TB, 0, stream>>>(bufB, b0, bufA, NN * DH, DH);

  // 5) GIN layer 1: acc = h + sum_neighbors(h); h = relu(acc @ Wg1 + bg1)
  k_copy4<<<blks((long long)NN * DH / 4, TB), TB, 0, stream>>>((const float4*)bufA,
                                                               (float4*)bufB, NN * DH / 4);
  k_gin_scatter<<<blks((long long)NE * 32, TB), TB, 0, stream>>>(esrc, edst, bufA, bufB);
  k_gemm_bf16<<<dim3(NN / 16, DH / 128), TB, 0, stream>>>(bufB, Wg1b, bg1, bufA,
                                                          NN, DH, DH, 1);

  // 6) GIN layer 2
  k_copy4<<<blks((long long)NN * DH / 4, TB), TB, 0, stream>>>((const float4*)bufA,
                                                               (float4*)bufB, NN * DH / 4);
  k_gin_scatter<<<blks((long long)NE * 32, TB), TB, 0, stream>>>(esrc, edst, bufA, bufB);
  k_gemm_bf16<<<dim3(NN / 16, DH / 128), TB, 0, stream>>>(bufB, Wg2b, bg2, bufA,
                                                          NN, DH, DH, 1);

  // 7) global mean pool
  k_fill<<<blks(NG * DH, TB), TB, 0, stream>>>(spool, 0.0f, NG * DH);
  k_fill<<<blks(NG, TB), TB, 0, stream>>>(cnt, 0.0f, NG);
  k_pool_scatter<<<blks((long long)NN * 32, TB), TB, 0, stream>>>(bufA, batch, spool, cnt);
  k_pool_mean<<<blks(NG * DH, TB), TB, 0, stream>>>(spool, cnt, mbuf);

  // 8) MLP head: g = relu(m @ Wh1 + bh1); out = g @ Wh2 + bh2
  k_gemm_bf16<<<dim3(NG / 16, DHID / 128), TB, 0, stream>>>(mbuf, Wh1b, bh1, gbuf,
                                                            NG, DH, DHID, 1);
  k_gemm_bf16<<<dim3(NG / 16, DOUT / 128), TB, 0, stream>>>(gbuf, Wh2b, bh2, (float*)d_out,
                                                            NG, DHID, DOUT, 0);
}
